// COVER_38972533243929
// MI455X (gfx1250) — matrix operand order repaired and checked by reading, not compiled, not run on hardware
//
#include <hip/hip_runtime.h>

typedef __attribute__((ext_vector_type(16))) _Float16 v16h;
typedef __attribute__((ext_vector_type(8)))  float    v8f;

#define AMPP   2
#define KSZ    5
#define NOFF   25
#define NB     4
#define NC     32
#define NH     256
#define NW     256
#define HEADS  4
#define HDIM   8
#define KROWS  12          // 8 rows + 2*AMP halo
#define KCOLS  20          // 16 cols + 2*AMP halo
#define KTILE  (NC*KROWS*KCOLS)   // 7680 floats = 30 KB LDS

__global__ __launch_bounds__(256)
void natt5x5_kernel(const float* __restrict__ xq,
                    const float* __restrict__ xk,
                    float* __restrict__ out) {
    __shared__ float kT[KTILE];

    const int tid = threadIdx.x;
    const int bx  = blockIdx.x % (NW/16);
    const int by  = (blockIdx.x / (NW/16)) % (NH/8);
    const int b   = blockIdx.x / ((NW/16)*(NH/8));
    const int x0  = bx*16, y0 = by*8;

    const float* qimg = xq + (size_t)b*NC*NH*NW;
    const float* kimg = xk + (size_t)b*NC*NH*NW;
    __builtin_prefetch(qimg + (size_t)y0*NW + x0, 0, 0);   // global_prefetch_b8

    // ---- stage k halo tile into LDS (coalesced along x) ----
    #pragma unroll
    for (int it = 0; it < KTILE/256; ++it) {
        int e   = tid + it*256;
        int c   = e / (KROWS*KCOLS);
        int rem = e % (KROWS*KCOLS);
        int row = rem / KCOLS, col = rem % KCOLS;
        int gy  = y0 - AMPP + row, gx = x0 - AMPP + col;
        float v = 0.0f;
        if ((unsigned)gy < NH && (unsigned)gx < NW)
            v = kimg[((size_t)c*NH + gy)*NW + gx];
        kT[e] = v;
    }
    __syncthreads();

    // ---- per-lane setup: wave32, lane pair (m, m+16) owns pixel m ----
    const int lane = tid & 31;
    const int wv   = tid >> 5;         // wave id -> row in tile
    const int l16  = lane & 15;
    const int hi   = lane >> 4;
    const int y    = y0 + wv, x = x0 + l16;

    // A-matrix 16-bit 16x32 slot map: half t -> K slot
    int  ldsoff[16];
    bool realf[16], validf[16];
    #pragma unroll
    for (int t = 0; t < 16; ++t) {
        int slot = t + (t < 8 ? (hi ? 8 : 0) : (hi ? 16 : 8));
        bool re  = slot < NOFF;
        int dy   = re ? (slot / KSZ - AMPP) : 0;
        int dx   = re ? (slot % KSZ - AMPP) : 0;
        realf[t]  = re;
        validf[t] = re && ((unsigned)(y + dy) < NH) && ((unsigned)(x + dx) < NW);
        ldsoff[t] = (wv + dy + AMPP)*KCOLS + (l16 + dx + AMPP);
    }

    const float scale = 0.35355339059327373f;   // 8^-0.5
    const float* qpix = qimg + (size_t)y*NW + x;

    float wavg[16];
    #pragma unroll
    for (int t = 0; t < 16; ++t) wavg[t] = 0.0f;

    // ---- per head: 25 dots (len 8) + softmax, accumulate head-mean ----
    #pragma unroll
    for (int h = 0; h < HEADS; ++h) {
        float s[16];
        #pragma unroll
        for (int t = 0; t < 16; ++t) s[t] = 0.0f;
        #pragma unroll
        for (int d = 0; d < HDIM; ++d) {
            int c = d*HEADS + h;                       // reshape(B, hd, heads, H, W)
            float qv = qpix[(size_t)c*NH*NW] * scale;
            const float* kc = kT + c*(KROWS*KCOLS);
            #pragma unroll
            for (int t = 0; t < 16; ++t)
                s[t] = fmaf(qv, kc[ldsoff[t]], s[t]);
        }
        float mx = -1.0e30f;
        #pragma unroll
        for (int t = 0; t < 16; ++t) {
            if (!validf[t]) s[t] = -1000.0f;           // reference OOB mask
            mx = fmaxf(mx, s[t]);
        }
        mx = fmaxf(mx, __shfl_xor(mx, 16, 32));        // pair-reduce (wave32)

        float e[16], sum = 0.0f;
        #pragma unroll
        for (int t = 0; t < 16; ++t) {
            float ev = realf[t] ? __expf(s[t] - mx) : 0.0f;
            e[t] = ev; sum += ev;
        }
        sum += __shfl_xor(sum, 16, 32);
        float inv = 0.25f / sum;                       // head-mean folded in
        #pragma unroll
        for (int t = 0; t < 16; ++t)
            wavg[t] = fmaf(e[t], inv, wavg[t]);
    }

    // ---- A: head-averaged softmax weights in 16x32 f16 layout ----
    v16h A;
    #pragma unroll
    for (int t = 0; t < 16; ++t) A[t] = (_Float16)wavg[t];

    // ---- B: 32x16 offset-grid; lane c<16 holds col c rows 0..7,16..23, lane c+16 rows 8..15,24..31 ----
    v16h Bm;
    {
        const int col = l16, base = hi * 16;
        #pragma unroll
        for (int t = 0; t < 16; ++t) {
            int k = t + (t < 8 ? (hi ? 8 : 0) : (hi ? 16 : 8));
            float v = 0.0f;
            if (k < NOFF) {
                if (col == 0)      v = (float)(k / KSZ - AMPP);   // E[dy]
                else if (col == 1) v = (float)(k % KSZ - AMPP);   // E[dx]
            }
            Bm[t] = (_Float16)v;
        }
    }

    v8f C = {};
    v8f D = __builtin_amdgcn_wmma_f32_16x16x32_f16(
        /*neg_a=*/false, A, /*neg_b=*/false, Bm,
        /*c_mod=*/(short)0, C, /*reuse_a=*/false, /*reuse_b=*/false);

    // ---- D layout: VGPR r, lanes0-15 -> (M=r, N=lane); lanes16-31 -> (M=r+8, N=lane-16)
    // lane 0/1 hold comp0/comp1 for pixels 0..7; lanes 16/17 for pixels 8..15.
    const int comp = l16;
    if (comp < 2) {
        float* ob = out + (((size_t)b*2 + comp)*NH + y)*NW + (x0 + hi*8);
        #pragma unroll
        for (int r = 0; r < 8; ++r) ob[r] = D[r];
    }
}

extern "C" void kernel_launch(void* const* d_in, const int* in_sizes, int n_in,
                              void* d_out, int out_size, void* d_ws, size_t ws_size,
                              hipStream_t stream) {
    const float* xq = (const float*)d_in[0];
    const float* xk = (const float*)d_in[1];
    (void)in_sizes; (void)n_in; (void)d_ws; (void)ws_size; (void)out_size;
    float* out = (float*)d_out;
    const int blocks = NB * (NH/8) * (NW/16);   // 4*32*16 = 2048
    natt5x5_kernel<<<blocks, 256, 0, stream>>>(xq, xk, out);
}
